// Attention_52982716563627
// MI455X (gfx1250) — matrix operand order repaired and checked by reading, not compiled, not run on hardware
//
#include <hip/hip_runtime.h>
#include <stdint.h>

// ---------------------------------------------------------------------------
// CDNA5 (gfx1250) fused attention: QKV gemm -> head-softmax attention -> proj
// All GEMMs via v_wmma_f32_16x16x32_bf16 (wave32, 16x16 tiles).
// bf16 conversion via native casts (hardware v_cvt path), global_prefetch_b8
// for next-tile readahead.
// ---------------------------------------------------------------------------

typedef __attribute__((ext_vector_type(16))) __bf16 bf16x16;
typedef __attribute__((ext_vector_type(8)))  float  floatx8;

struct alignas(16) U128 { unsigned int w[4]; };

union Frag {
    bf16x16 v;
    U128    q[2];
};

__device__ __forceinline__ __bf16 f2bf(float f) { return (__bf16)f; }

#define WMMA_BF16(A, B, C) \
    __builtin_amdgcn_wmma_f32_16x16x32_bf16(false, (A), false, (B), (short)0, (C), false, false)

// problem constants
#define BATCH 2
#define SEQ   2048
#define EMB   1024
#define HEADS 16
#define HD    64

// ---------------------------------------------------------------------------
// Kernel 1: qkv = x @ W_atten + b_atten, scattered to Q [B,H,S,64] bf16,
// K [B,H,S,64] bf16, and V transposed Vt [B,H,64,S] bf16.
// 64x64 output tile per block, 256 threads (8 waves), K-step 32.
// ---------------------------------------------------------------------------
__global__ __launch_bounds__(256) void qkv_gemm_kernel(
    const float* __restrict__ x, const float* __restrict__ W,
    const float* __restrict__ bias,
    __bf16* __restrict__ Qb, __bf16* __restrict__ Kb,
    __bf16* __restrict__ Vt)
{
    __shared__ __bf16 lds_a[64 * 40];   // [m][k], row padded to 40 (80B)
    __shared__ __bf16 lds_w[64 * 40];   // [n][k] (transposed), padded

    const int tid  = threadIdx.x;
    const int lane = tid & 31;
    const int w    = tid >> 5;
    const int wm   = w >> 1;        // 0..3 : 16-row slab
    const int wn   = w & 1;         // 0..1 : 32-col slab
    const int m0   = blockIdx.y * 64;
    const int n0   = blockIdx.x * 64;

    const int r   = (lane < 16) ? lane : lane - 16;
    const int akb = (lane < 16) ? 0 : 16;   // A-frag K base
    const int bkb = (lane < 16) ? 0 : 16;   // B-frag K base

    const int arow = tid >> 2, akg = (tid & 3) * 8;     // A stager coords
    const int wrow = tid >> 3, wng = (tid & 7) * 8;     // W stager coords

    floatx8 acc[2];
    acc[0] = (floatx8){0,0,0,0,0,0,0,0};
    acc[1] = (floatx8){0,0,0,0,0,0,0,0};

    for (int kk = 0; kk < EMB; kk += 32) {
        // readahead next K-step (global_prefetch_b8)
        if (kk + 32 < EMB) {
            __builtin_prefetch(x + (size_t)(m0 + arow) * EMB + kk + 32 + akg, 0, 1);
            __builtin_prefetch(W + (size_t)(kk + 32 + wrow) * (3 * EMB) + n0 + wng, 0, 1);
        }
        // stage A tile 64x32, f32 -> bf16
        {
            const float* src = x + (size_t)(m0 + arow) * EMB + kk + akg;
            __bf16* dst = &lds_a[arow * 40 + akg];
            #pragma unroll
            for (int i = 0; i < 8; ++i) dst[i] = f2bf(src[i]);
        }
        // stage W tile 32x64 transposed: lds_w[n][k]
        {
            const float* src = W + (size_t)(kk + wrow) * (3 * EMB) + n0 + wng;
            #pragma unroll
            for (int i = 0; i < 8; ++i) lds_w[(wng + i) * 40 + wrow] = f2bf(src[i]);
        }
        __syncthreads();

        Frag a;
        {
            const U128* p = (const U128*)&lds_a[(wm * 16 + r) * 40 + akb];
            a.q[0] = p[0]; a.q[1] = p[1];           // 16 contiguous K values
        }
        #pragma unroll
        for (int t = 0; t < 2; ++t) {
            Frag bf;
            const U128* p = (const U128*)&lds_w[(wn * 32 + t * 16 + r) * 40 + bkb];
            bf.q[0] = p[0]; bf.q[1] = p[1];         // 16 contiguous K values
            acc[t] = WMMA_BF16(a.v, bf.v, acc[t]);
        }
        __syncthreads();
    }

    // epilogue: add bias, scatter into Q / K / Vt
    #pragma unroll
    for (int t = 0; t < 2; ++t) {
        const int col = wn * 32 + t * 16 + r;
        const int n   = n0 + col;
        const int h   = n / 192, j = n % 192;       // reshape (H, 3*hd)
        const float bv = bias[n];
        #pragma unroll
        for (int i = 0; i < 8; ++i) {
            const int row = wm * 16 + ((lane < 16) ? i : i + 8);
            const int m = m0 + row;
            const int b = m >> 11, s = m & (SEQ - 1);
            const __bf16 val = f2bf(acc[t][i] + bv);
            const int bh = b * HEADS + h;
            if (j < 64) {
                Qb[((size_t)bh * SEQ + s) * HD + j] = val;
            } else if (j < 128) {
                Kb[((size_t)bh * SEQ + s) * HD + (j - 64)] = val;
            } else {
                Vt[((size_t)bh * HD + (j - 128)) * SEQ + s] = val;   // transposed
            }
        }
    }
}

// ---------------------------------------------------------------------------
// Kernel 2: fused scores -> head-axis softmax (w/ +1e5 causal fill) -> P@V.
// One block = one (b, 16-row q tile). 512 threads = 16 waves = 16 heads.
// Iterates over 32-key tiles; scores never leave LDS.
// ---------------------------------------------------------------------------
__global__ __launch_bounds__(512) void attn_kernel(
    const __bf16* __restrict__ Qb, const __bf16* __restrict__ Kb,
    const __bf16* __restrict__ Vt, __bf16* __restrict__ Ab)
{
    __shared__ float  lds_s[512 * 17];        // [pos=q*32+k][head], padded
    __shared__ __bf16 lds_p[HEADS * 16 * 40]; // [h][q][k] bf16, row padded

    const int tid  = threadIdx.x;
    const int lane = tid & 31;
    const int h    = tid >> 5;                       // wave id == head id
    const int b    = blockIdx.x >> 7;
    const int q0   = (blockIdx.x & 127) * 16;

    const int r   = (lane < 16) ? lane : lane - 16;
    const int akb = (lane < 16) ? 0 : 16;
    const int bkb = (lane < 16) ? 0 : 16;

    const size_t headQK = (size_t)(b * HEADS + h) * SEQ * HD;
    const size_t headV  = (size_t)(b * HEADS + h) * HD * SEQ;

    // Q A-fragments, K-dim d: frag j covers d = 32j..32j+31 (reused all k-tiles)
    Frag aq[2];
    #pragma unroll
    for (int j = 0; j < 2; ++j) {
        const U128* p = (const U128*)(Qb + headQK + (size_t)(q0 + r) * HD + j * 32 + akb);
        aq[j].q[0] = p[0]; aq[j].q[1] = p[1];
    }

    floatx8 acc[4];
    #pragma unroll
    for (int nt = 0; nt < 4; ++nt) acc[nt] = (floatx8){0,0,0,0,0,0,0,0};

    for (int kt = 0; kt < SEQ / 32; ++kt) {
        const int k0 = kt * 32;

        // readahead next k-tile K and V rows (global_prefetch_b8)
        if (k0 + 32 < SEQ) {
            __builtin_prefetch(Kb + headQK + (size_t)(k0 + 32 + r) * HD + bkb, 0, 1);
            __builtin_prefetch(Vt + headV + (size_t)r * SEQ + k0 + 32 + bkb, 0, 1);
        }

        // ---- scores: s[h] = q @ k^T for 16q x 32k, two 16x16 tiles ----
        #pragma unroll
        for (int t = 0; t < 2; ++t) {
            floatx8 c = (floatx8){0,0,0,0,0,0,0,0};
            #pragma unroll
            for (int j = 0; j < 2; ++j) {
                Frag kf;
                const U128* p = (const U128*)(Kb + headQK +
                        (size_t)(k0 + t * 16 + r) * HD + j * 32 + bkb);
                kf.q[0] = p[0]; kf.q[1] = p[1];
                c = WMMA_BF16(aq[j].v, kf.v, c);
            }
            #pragma unroll
            for (int i = 0; i < 8; ++i) {
                const int row = (lane < 16) ? i : i + 8;
                const int col = t * 16 + r;
                lds_s[(row * 32 + col) * 17 + h] = c[i];
            }
        }
        __syncthreads();

        // ---- softmax over the 16 heads at one (q,k) position per thread ----
        {
            const int pos = tid;                    // 512 positions exactly
            const int row = pos >> 5, kcol = pos & 31;
            const int qg = q0 + row, kg = k0 + kcol;
            if (kg > qg) {
                // masked: all heads filled with +1e5 -> uniform 1/16
                #pragma unroll
                for (int i2 = 0; i2 < HEADS; ++i2)
                    lds_p[(i2 * 16 + row) * 40 + kcol] = f2bf(0.0625f);
            } else {
                float sv[HEADS];
                float mx = -3.0e38f;
                #pragma unroll
                for (int i2 = 0; i2 < HEADS; ++i2) {
                    sv[i2] = lds_s[pos * 17 + i2];
                    mx = fmaxf(mx, sv[i2]);
                }
                float sum = 0.f;
                #pragma unroll
                for (int i2 = 0; i2 < HEADS; ++i2) {
                    sv[i2] = __expf(sv[i2] - mx);
                    sum += sv[i2];
                }
                const float inv = 1.0f / sum;
                #pragma unroll
                for (int i2 = 0; i2 < HEADS; ++i2)
                    lds_p[(i2 * 16 + row) * 40 + kcol] = f2bf(sv[i2] * inv);
            }
        }
        __syncthreads();

        // ---- P @ V: 16x32 @ 32x64 -> accumulate 16x64 per head ----
        Frag pa;
        {
            const U128* p = (const U128*)&lds_p[(h * 16 + r) * 40 + akb];
            pa.q[0] = p[0]; pa.q[1] = p[1];
        }
        #pragma unroll
        for (int nt = 0; nt < 4; ++nt) {
            Frag vf;
            const U128* p = (const U128*)(Vt + headV +
                    (size_t)(nt * 16 + r) * SEQ + k0 + bkb);
            vf.q[0] = p[0]; vf.q[1] = p[1];
            acc[nt] = WMMA_BF16(pa.v, vf.v, acc[nt]);
        }
        __syncthreads();
    }

    // epilogue: a[b][s][h*64+d] bf16
    #pragma unroll
    for (int nt = 0; nt < 4; ++nt) {
        #pragma unroll
        for (int i = 0; i < 8; ++i) {
            const int row = (lane < 16) ? i : i + 8;
            const int d = nt * 16 + r;
            const int s = q0 + row;
            Ab[((size_t)(b * SEQ + s)) * EMB + h * HD + d] = f2bf(acc[nt][i]);
        }
    }
}

// ---------------------------------------------------------------------------
// Kernel 3: out = a @ W_proj + b_proj (f32 out). A already bf16 in workspace.
// ---------------------------------------------------------------------------
__global__ __launch_bounds__(256) void proj_gemm_kernel(
    const __bf16* __restrict__ A, const float* __restrict__ W,
    const float* __restrict__ bias, float* __restrict__ out)
{
    __shared__ __bf16 lds_a[64 * 40];
    __shared__ __bf16 lds_w[64 * 40];

    const int tid  = threadIdx.x;
    const int lane = tid & 31;
    const int w    = tid >> 5;
    const int wm   = w >> 1;
    const int wn   = w & 1;
    const int m0   = blockIdx.y * 64;
    const int n0   = blockIdx.x * 64;

    const int r   = (lane < 16) ? lane : lane - 16;
    const int akb = (lane < 16) ? 0 : 16;
    const int bkb = (lane < 16) ? 0 : 16;

    const int arow = tid >> 2, akg = (tid & 3) * 8;
    const int wrow = tid >> 3, wng = (tid & 7) * 8;

    floatx8 acc[2];
    acc[0] = (floatx8){0,0,0,0,0,0,0,0};
    acc[1] = (floatx8){0,0,0,0,0,0,0,0};

    for (int kk = 0; kk < EMB; kk += 32) {
        if (kk + 32 < EMB) {
            __builtin_prefetch(A + (size_t)(m0 + arow) * EMB + kk + 32 + akg, 0, 1);
            __builtin_prefetch(W + (size_t)(kk + 32 + wrow) * EMB + n0 + wng, 0, 1);
        }
        {   // stage A (already bf16): 16B vector copies
            const U128* src = (const U128*)(A + (size_t)(m0 + arow) * EMB + kk + akg);
            *(U128*)&lds_a[arow * 40 + akg] = src[0];
        }
        {   // stage W transposed, f32 -> bf16
            const float* src = W + (size_t)(kk + wrow) * EMB + n0 + wng;
            #pragma unroll
            for (int i = 0; i < 8; ++i) lds_w[(wng + i) * 40 + wrow] = f2bf(src[i]);
        }
        __syncthreads();

        Frag a;
        {
            const U128* p = (const U128*)&lds_a[(wm * 16 + r) * 40 + akb];
            a.q[0] = p[0]; a.q[1] = p[1];
        }
        #pragma unroll
        for (int t = 0; t < 2; ++t) {
            Frag bf;
            const U128* p = (const U128*)&lds_w[(wn * 32 + t * 16 + r) * 40 + bkb];
            bf.q[0] = p[0]; bf.q[1] = p[1];
            acc[t] = WMMA_BF16(a.v, bf.v, acc[t]);
        }
        __syncthreads();
    }

    #pragma unroll
    for (int t = 0; t < 2; ++t) {
        const int col = wn * 32 + t * 16 + r;
        const int n = n0 + col;
        const float bv = bias[n];
        #pragma unroll
        for (int i = 0; i < 8; ++i) {
            const int row = wm * 16 + ((lane < 16) ? i : i + 8);
            out[(size_t)(m0 + row) * EMB + n] = acc[t][i] + bv;
        }
    }
}

// ---------------------------------------------------------------------------
extern "C" void kernel_launch(void* const* d_in, const int* in_sizes, int n_in,
                              void* d_out, int out_size, void* d_ws, size_t ws_size,
                              hipStream_t stream) {
    (void)in_sizes; (void)n_in; (void)out_size; (void)ws_size;
    const float* x       = (const float*)d_in[0];
    const float* W_atten = (const float*)d_in[1];
    const float* b_atten = (const float*)d_in[2];
    const float* W_proj  = (const float*)d_in[3];
    const float* b_proj  = (const float*)d_in[4];
    float* out = (float*)d_out;

    char* ws = (char*)d_ws;
    const size_t SZ = (size_t)BATCH * HEADS * SEQ * HD * sizeof(__bf16); // 8 MiB
    __bf16* Qb = (__bf16*)(ws);
    __bf16* Kb = (__bf16*)(ws + SZ);
    __bf16* Vt = (__bf16*)(ws + 2 * SZ);
    __bf16* Ab = (__bf16*)(ws + 3 * SZ);

    // QKV: M = B*S = 4096, N = 3E = 3072
    qkv_gemm_kernel<<<dim3(3 * EMB / 64, BATCH * SEQ / 64), 256, 0, stream>>>(
        x, W_atten, b_atten, Qb, Kb, Vt);
    // Attention: one block per (b, 16-row q tile)
    attn_kernel<<<dim3(BATCH * SEQ / 16), 512, 0, stream>>>(Qb, Kb, Vt, Ab);
    // Projection: M = 4096, N = 1024
    proj_gemm_kernel<<<dim3(EMB / 64, BATCH * SEQ / 64), 256, 0, stream>>>(
        Ab, W_proj, b_proj, out);
}
